// DeepSpeedMLP_37125697307027
// MI455X (gfx1250) — matrix operand order repaired and checked by reading, not compiled, not run on hardware
//
#include <hip/hip_runtime.h>
#include <hip/hip_bf16.h>

typedef __attribute__((ext_vector_type(16))) __bf16 v16bf;
typedef __attribute__((ext_vector_type(8)))  __bf16 v8bf;
typedef __attribute__((ext_vector_type(8)))  float  v8f;

#define HD   2048
#define IDIM 8192
#define NTOK 8192        // B*S = 4*2048
#define LN_EPS 1e-5f

// low 32 bits of a flat pointer to LDS == LDS byte offset (ISA 10.2)
#define LDS_OFF(p) ((unsigned)(size_t)(p))

// ---------------------------------------------------------------------------
// fp32 [K][N] row-major  ->  bf16 [N][K] (transposed), 32x32 LDS tiles
// ---------------------------------------------------------------------------
__global__ __launch_bounds__(256)
void cvt_t_bf16(const float* __restrict__ src, __bf16* __restrict__ dst,
                int K, int N)
{
    __shared__ float tile[32][33];
    const int tid = threadIdx.x;
    const int tx  = tid & 31;
    const int ty  = tid >> 5;           // 0..7
    const int kb  = blockIdx.y * 32;
    const int nb  = blockIdx.x * 32;

#pragma unroll
    for (int r = 0; r < 4; ++r)
        tile[ty + r * 8][tx] = src[(size_t)(kb + ty + r * 8) * N + nb + tx];
    __syncthreads();
#pragma unroll
    for (int r = 0; r < 4; ++r)
        dst[(size_t)(nb + ty + r * 8) * K + kb + tx] =
            (__bf16)tile[tx][ty + r * 8];
}

// ---------------------------------------------------------------------------
// x = input + bias + residual ; LayerNorm(x)*ln_w + attn_nb -> bf16 h
// one 256-thread block per token row (H = 2048, 8 elems/thread)
// ---------------------------------------------------------------------------
__global__ __launch_bounds__(256)
void ln_fuse(const float* __restrict__ inp, const float* __restrict__ res,
             const float* __restrict__ bias, const float* __restrict__ lnw,
             const float* __restrict__ beta, __bf16* __restrict__ hOut)
{
    __shared__ float s1[256];
    __shared__ float s2[256];

    const int row  = blockIdx.x;
    const int tid  = threadIdx.x;
    const int base = tid * 8;

    const float* pi = inp + (size_t)row * HD;
    const float* pr = res + (size_t)row * HD;

    float x[8];
#pragma unroll
    for (int j = 0; j < 8; j += 4) {
        float4 vi = *(const float4*)(pi + base + j);
        float4 vr = *(const float4*)(pr + base + j);
        float4 vb = *(const float4*)(bias + base + j);
        x[j + 0] = vi.x + vr.x + vb.x;
        x[j + 1] = vi.y + vr.y + vb.y;
        x[j + 2] = vi.z + vr.z + vb.z;
        x[j + 3] = vi.w + vr.w + vb.w;
    }

    float a1 = 0.f, a2 = 0.f;
#pragma unroll
    for (int j = 0; j < 8; ++j) { a1 += x[j]; a2 += x[j] * x[j]; }
    s1[tid] = a1;
    s2[tid] = a2;
    __syncthreads();
#pragma unroll
    for (int off = 128; off > 0; off >>= 1) {
        if (tid < off) { s1[tid] += s1[tid + off]; s2[tid] += s2[tid + off]; }
        __syncthreads();
    }
    const float mu  = s1[0] * (1.f / HD);
    const float var = s2[0] * (1.f / HD) - mu * mu;
    const float rs  = rsqrtf(var + LN_EPS);

    __bf16* po = hOut + (size_t)row * HD + base;
#pragma unroll
    for (int j = 0; j < 8; ++j) {
        float h = (x[j] - mu) * rs * lnw[base + j] + beta[base + j];
        po[j] = (__bf16)h;
    }
}

// ---------------------------------------------------------------------------
// bf16 WMMA GEMM: C[M,NC] = A[M,KC] * Bt[NC,KC]^T (+ biasN).
// A row-major, Bt row-major (pre-transposed weights). All strides constexpr.
// 256 threads = 8 waves; WG tile 128x128; wave tile 32x64 (2x4 of 16x16).
// K-step 64, double-buffered LDS, 2-deep async pipeline; last step peeled.
// MODE 0: relu(acc + biasN)              -> bf16 out
// MODE 1: acc + biasN + (xin+xres+xbias) -> f32 out
// ---------------------------------------------------------------------------
template <int NC, int KC, int MODE>
__global__ __launch_bounds__(256)
void wmma_gemm(const __bf16* __restrict__ A, const __bf16* __restrict__ Bt,
               const float* __restrict__ biasN,
               __bf16* __restrict__ outBf, float* __restrict__ outF,
               const float* __restrict__ xin, const float* __restrict__ xres,
               const float* __restrict__ xbias)
{
    __shared__ __align__(32) __bf16 lsA[2][128 * 64];   // 2 x 16 KB (M x K panel)
    __shared__ __align__(32) __bf16 lsB[2][128 * 64];   // 2 x 16 KB (N x K panel)

    const int tid   = threadIdx.x;
    const int lane  = tid & 31;
    const int wave  = tid >> 5;    // 0..7
    const int waveM = wave >> 1;   // 0..3  (rows of 32)
    const int waveN = wave & 1;    // 0..1  (cols of 64)

    const int mBase = blockIdx.y * 128;
    const int nBase = blockIdx.x * 128;

    v8f acc[2][4];
#pragma unroll
    for (int i = 0; i < 2; ++i)
#pragma unroll
        for (int j = 0; j < 4; ++j)
#pragma unroll
            for (int r = 0; r < 8; ++r) acc[i][j][r] = 0.f;

    // staging: 128 rows x 64 cols bf16 = 1024 x 16B chunks; 4 chunks/thread/array
    const int rowB = tid >> 3;           // 0..31 (+32*c)
    const int col8 = (tid & 7) * 8;      // 0..56

    // running global pointers, advanced 64 elements (128 B) per stage
    const __bf16* gA[4];
    const __bf16* gB[4];
#pragma unroll
    for (int c = 0; c < 4; ++c) {
        gA[c] = &A [(size_t)(mBase + rowB + c * 32) * KC + col8];
        gB[c] = &Bt[(size_t)(nBase + rowB + c * 32) * KC + col8];
    }

    // issue one K-panel (8 async transfers / thread) into buffer `buf`
    auto stage = [&](int buf) {
#pragma unroll
        for (int c = 0; c < 4; ++c) {
            const int row = rowB + c * 32;
            const unsigned la = LDS_OFF(&lsA[buf][row * 64 + col8]);
            const unsigned lb = LDS_OFF(&lsB[buf][row * 64 + col8]);
            asm volatile("global_load_async_to_lds_b128 %0, %1, off"
                         :: "v"(la), "v"(gA[c]) : "memory");
            asm volatile("global_load_async_to_lds_b128 %0, %1, off"
                         :: "v"(lb), "v"(gB[c]) : "memory");
            gA[c] += 64;
            gB[c] += 64;
        }
    };

    const int lr = lane & 15;
    const int hi = lane >> 4;            // 0 or 1
    const int mo = hi * 8;

    auto compute = [&](int buf) {
#pragma unroll
        for (int kk = 0; kk < 2; ++kk) {
            // A fragments (16-bit A 16x32, contiguous k order shared with B):
            // lanes 0-15: K=0..15; lanes 16-31: K=16..31
            v16bf afrag[2];
#pragma unroll
            for (int i = 0; i < 2; ++i) {
                const __bf16* ap =
                    &lsA[buf][(waveM * 32 + i * 16 + lr) * 64 + kk * 32 + hi * 16];
                v8bf lo = *(const v8bf*)ap;
                v8bf hh = *(const v8bf*)(ap + 8);
                afrag[i] = __builtin_shufflevector(lo, hh,
                    0, 1, 2, 3, 4, 5, 6, 7, 8, 9, 10, 11, 12, 13, 14, 15);
            }
            // B fragments: lanes 0-15: N=lr, K=0..15; lanes 16-31: K=16..31
            v16bf bfrag[4];
#pragma unroll
            for (int j = 0; j < 4; ++j) {
                const __bf16* bp =
                    &lsB[buf][(waveN * 64 + j * 16 + lr) * 64 + kk * 32 + hi * 16];
                bfrag[j] = *(const v16bf*)bp;
            }
#pragma unroll
            for (int i = 0; i < 2; ++i)
#pragma unroll
                for (int j = 0; j < 4; ++j)
                    acc[i][j] = __builtin_amdgcn_wmma_f32_16x16x32_bf16(
                        false, afrag[i], false, bfrag[j],
                        (short)0, acc[i][j], false, false);
        }
    };

    constexpr int kSteps = KC >> 6;      // KT = 64; kSteps >= 2
    stage(0);                            // prologue fill

    // steady state: unconditional issue-ahead pipeline (last step peeled)
#pragma unroll 1
    for (int ks = 0; ks < kSteps - 1; ++ks) {
        const int cur = ks & 1;
        stage(cur ^ 1);
        // async completes in order: <=8 pending -> previous group landed
        asm volatile("s_wait_asynccnt 0x8" ::: "memory");
        __syncthreads();   // all waves' `cur` tiles resident
        compute(cur);
        __syncthreads();   // `cur` fully consumed -> safe to refill
    }
    asm volatile("s_wait_asynccnt 0x0" ::: "memory");
    __syncthreads();
    compute((kSteps - 1) & 1);

    // epilogue: C/D layout - lane: n = lane&15, m = (lane>>4)*8 + vgpr index
#pragma unroll
    for (int i = 0; i < 2; ++i) {
#pragma unroll
        for (int j = 0; j < 4; ++j) {
            const int nG = nBase + waveN * 64 + j * 16 + lr;
            const float bv = biasN[nG];
#pragma unroll
            for (int r = 0; r < 8; ++r) {
                const int mG = mBase + waveM * 32 + i * 16 + mo + r;
                float v = acc[i][j][r] + bv;
                if constexpr (MODE == 0) {
                    v = v > 0.f ? v : 0.f;
                    outBf[(size_t)mG * NC + nG] = (__bf16)v;
                } else {
                    const size_t off = (size_t)mG * NC + nG;
                    const float xv = xin[off] + xres[off] + xbias[nG];
                    outF[off] = v + xv;
                }
            }
        }
    }
}

// ---------------------------------------------------------------------------
extern "C" void kernel_launch(void* const* d_in, const int* in_sizes, int n_in,
                              void* d_out, int out_size, void* d_ws, size_t ws_size,
                              hipStream_t stream)
{
    const float* input    = (const float*)d_in[0];
    const float* residual = (const float*)d_in[1];
    // d_in[2] residual_norm: unused by reference
    const float* bias     = (const float*)d_in[3];
    // d_in[4] weight: unused by reference
    const float* ln_w     = (const float*)d_in[5];
    const float* attn_nb  = (const float*)d_in[6];
    const float* inter_w  = (const float*)d_in[7];
    const float* inter_b  = (const float*)d_in[8];
    const float* output_w = (const float*)d_in[9];
    const float* output_b = (const float*)d_in[10];
    float* out = (float*)d_out;

    const size_t MB32 = 33554432u; // 32 MiB
    char* ws = (char*)d_ws;
    __bf16* hBf     = (__bf16*)(ws + 0 * MB32); // [NTOK,HD]   bf16
    __bf16* wInterT = (__bf16*)(ws + 1 * MB32); // [IDIM,HD]   bf16 (inter_w^T)
    __bf16* h1Bf    = (__bf16*)(ws + 2 * MB32); // [NTOK,IDIM] bf16
    __bf16* wOutT   = (__bf16*)(ws + 3 * MB32); // [HD,IDIM]   bf16 (output_w^T)

    // weight convert + transpose: inter_w [HD,IDIM] -> [IDIM,HD]
    dim3 gt1(IDIM / 32, HD / 32);
    cvt_t_bf16<<<gt1, 256, 0, stream>>>(inter_w, wInterT, HD, IDIM);
    // output_w [IDIM,HD] -> [HD,IDIM]
    dim3 gt2(HD / 32, IDIM / 32);
    cvt_t_bf16<<<gt2, 256, 0, stream>>>(output_w, wOutT, IDIM, HD);

    ln_fuse<<<NTOK, 256, 0, stream>>>(input, residual, bias, ln_w, attn_nb, hBf);

    dim3 g1(IDIM / 128, NTOK / 128);
    wmma_gemm<IDIM, HD, 0><<<g1, 256, 0, stream>>>(
        hBf, wInterT, inter_b, h1Bf, nullptr, nullptr, nullptr, nullptr);

    dim3 g2(HD / 128, NTOK / 128);
    wmma_gemm<HD, IDIM, 1><<<g2, 256, 0, stream>>>(
        h1Bf, wOutT, output_b, nullptr, out, input, residual, bias);
}
